// DoubleConv_25211458027718
// MI455X (gfx1250) — matrix operand order repaired and checked by reading, not compiled, not run on hardware
//
#include <hip/hip_runtime.h>

// ---------------------------------------------------------------------------
// Problem constants (from the reference)
// ---------------------------------------------------------------------------
#define N_TILES 6
#define NX      96
#define NN      (N_TILES * NX * NX)   // 55296 nodes
#define BATCH   4
#define IN_C    128
#define HID_C   256

typedef __bf16 bf16;
typedef __attribute__((ext_vector_type(16))) bf16  v16bf;
typedef __attribute__((ext_vector_type(8)))  bf16  v8bf;
typedef __attribute__((ext_vector_type(8)))  float v8f;

static __device__ __forceinline__ float to_f(float x) { return x; }
static __device__ __forceinline__ float to_f(bf16 x)  { return (float)x; }

static __device__ __forceinline__ void store_val(float* p, float v) { *p = v; }
static __device__ __forceinline__ void store_val(bf16*  p, float v) { *p = (bf16)v; }

// ---------------------------------------------------------------------------
// Weight packing: combined K = [W_self ; W_neigh]  (2*Cin x 256), f32 -> bf16,
// laid out as Wp[(k/16)*256 + n][16] so a WMMA B-fragment lane reads 16
// contiguous bf16 (two global_load_b128), coalesced across the wave.
// ---------------------------------------------------------------------------
__global__ __launch_bounds__(256) void pack_w(const float* __restrict__ Ws,
                                              const float* __restrict__ Wn,
                                              int Cin, bf16* __restrict__ out) {
    int idx = blockIdx.x * 256 + threadIdx.x;
    int total = 2 * Cin * 256;
    if (idx >= total) return;
    int k = idx >> 8;          // combined K index
    int n = idx & 255;         // output channel
    float v = (k < Cin) ? Ws[k * 256 + n] : Wn[(k - Cin) * 256 + n];
    int g = k >> 4;            // 16-row group
    int t = k & 15;
    out[((long)(g << 8) + n) * 16 + t] = (bf16)v;
}

// ---------------------------------------------------------------------------
// Fused SAGE layer: out = relu([h | mean4(h_nbr)] * [Wself;Wneigh] + b)
// One workgroup (8 waves / 256 threads) = 32 nodes x 256 channels.
//   wave = (mt in 0..1) x (ng in 0..3); each wave: 16x16 tiles over 4 N-tiles.
// ---------------------------------------------------------------------------
template <int CIN, typename InT, typename OutT>
__global__ __launch_bounds__(256) void sage_layer(
    const InT* __restrict__ hin,    // [B, NN, CIN]
    const int*  __restrict__ nbrs,  // [NN, 4]
    const bf16* __restrict__ Wp,    // packed [(2*CIN)/16][256][16] bf16
    const float* __restrict__ bias, // [256]
    OutT* __restrict__ hout)        // [B, NN, 256]
{
    constexpr int K2 = 2 * CIN;
    __shared__ __align__(16) bf16 Ac[32 * K2];   // [node][self | neigh-mean]

    const int bb       = blockIdx.y;
    const int nodeBase = blockIdx.x * 32;
    const int tid      = threadIdx.x;
    const long inBase  = (long)bb * NN * CIN;

    // ---- stage A panel: self features + 4-neighbor mean, f32 -> bf16 ------
    for (int i = tid; i < 32 * CIN; i += 256) {
        int m = i / CIN;
        int c = i - m * CIN;
        int node = nodeBase + m;
        float self = to_f(hin[inBase + (long)node * CIN + c]);
        const int* nb = &nbrs[node * 4];
        float s = to_f(hin[inBase + (long)nb[0] * CIN + c])
                + to_f(hin[inBase + (long)nb[1] * CIN + c])
                + to_f(hin[inBase + (long)nb[2] * CIN + c])
                + to_f(hin[inBase + (long)nb[3] * CIN + c]);
        Ac[m * K2 + c]       = (bf16)self;
        Ac[m * K2 + CIN + c] = (bf16)(0.25f * s);
    }
    __syncthreads();

    // ---- WMMA compute ------------------------------------------------------
    const int lane = tid & 31;
    const int wave = tid >> 5;
    const int mt   = wave & 1;         // M-tile within the 32-node block
    const int ng   = wave >> 1;        // N-group (4 tiles each)
    const int lm   = lane & 15;
    const int lh   = lane >> 4;        // lane half selects K sub-range
    const int m    = mt * 16 + lm;
    const bf16* arow = &Ac[m * K2 + lh * 16];

    for (int nt = 0; nt < 4; ++nt) {
        const int n0 = (ng * 4 + nt) * 16;
        const int n  = n0 + lm;
        v8f acc = {};

#pragma unroll
        for (int kc = 0; kc < K2 / 32; ++kc) {
            // A fragment (contiguous K order to match B): lanes 0-15: K 0-15,
            // lanes 16-31: K 16-31 -> two ds_load_b128 per lane.
            v8bf alo = *(const v8bf*)(arow + kc * 32);
            v8bf ahi = *(const v8bf*)(arow + kc * 32 + 8);
            v16bf a;
#pragma unroll
            for (int t = 0; t < 8; ++t) { a[t] = alo[t]; a[t + 8] = ahi[t]; }

            // B fragment: lane (n, lh) holds K = (kc*2+lh)*16 .. +15 for
            // column n -> 32 contiguous bytes from the packed weights.
            const bf16* bp = Wp + ((long)((kc * 2 + lh) * 256 + n)) * 16;
            v8bf blo = *(const v8bf*)(bp);
            v8bf bhi = *(const v8bf*)(bp + 8);
            v16bf bfr;
#pragma unroll
            for (int t = 0; t < 8; ++t) { bfr[t] = blo[t]; bfr[t + 8] = bhi[t]; }

            acc = __builtin_amdgcn_wmma_f32_16x16x32_bf16(
                false, a, false, bfr, (short)0, acc, false, false);
        }

        const float bv = bias[n];
        // f32 C/D layout: VGPR r -> row r (lanes 0-15) / row r+8 (lanes 16-31),
        // column = lane & 15.
#pragma unroll
        for (int r = 0; r < 8; ++r) {
            int mm = r + (lh ? 8 : 0);
            float v = acc[r] + bv;
            v = v > 0.0f ? v : 0.0f;
            store_val(&hout[(long)bb * NN * 256 +
                            (long)(nodeBase + mt * 16 + mm) * 256 + n], v);
        }
    }
}

// ---------------------------------------------------------------------------
// Launcher: pack weights (bf16, fragment layout) -> layer1 (f32 in, bf16 h1
// in workspace) -> layer2 (bf16 in, f32 out).
// ---------------------------------------------------------------------------
extern "C" void kernel_launch(void* const* d_in, const int* in_sizes, int n_in,
                              void* d_out, int out_size, void* d_ws, size_t ws_size,
                              hipStream_t stream) {
    const float* x    = (const float*)d_in[0];
    const int*   nbr  = (const int*)  d_in[1];
    const float* Ws1  = (const float*)d_in[2];
    const float* Wn1  = (const float*)d_in[3];
    const float* b1   = (const float*)d_in[4];
    const float* Ws2  = (const float*)d_in[5];
    const float* Wn2  = (const float*)d_in[6];
    const float* b2   = (const float*)d_in[7];
    float*       out  = (float*)d_out;

    char* ws = (char*)d_ws;
    const size_t wp1_bytes = (size_t)2 * IN_C  * HID_C * sizeof(bf16); // 128 KB
    const size_t wp2_bytes = (size_t)2 * HID_C * HID_C * sizeof(bf16); // 256 KB
    bf16* Wp1 = (bf16*)ws;
    bf16* Wp2 = (bf16*)(ws + wp1_bytes);
    bf16* h1  = (bf16*)(ws + wp1_bytes + wp2_bytes);                   // [B,NN,256]

    pack_w<<<(2 * IN_C  * HID_C + 255) / 256, 256, 0, stream>>>(Ws1, Wn1, IN_C,  Wp1);
    pack_w<<<(2 * HID_C * HID_C + 255) / 256, 256, 0, stream>>>(Ws2, Wn2, HID_C, Wp2);

    dim3 grid(NN / 32, BATCH);
    sage_layer<IN_C,  float, bf16><<<grid, 256, 0, stream>>>(x,  nbr, Wp1, b1, h1);
    sage_layer<HID_C, bf16, float><<<grid, 256, 0, stream>>>(h1, nbr, Wp2, b2, out);
}
